// Tokenizer_69518340653130
// MI455X (gfx1250) — matrix operand order repaired and checked by reading, not compiled, not run on hardware
//
#include <hip/hip_runtime.h>
#include <stdint.h>

// ---------------------------------------------------------------------------
// VQ-VAE tokenizer forward on gfx1250 (MI455X):
//   bf16 WMMA GEMMs with double-buffered GLOBAL_LOAD_ASYNC_TO_LDS staging
//   (branch-free steady state) + fused codebook argmin via WMMA.
// ---------------------------------------------------------------------------

typedef __attribute__((ext_vector_type(16))) __bf16 v16bf;
typedef __attribute__((ext_vector_type(8)))  float  v8f;

constexpr int NB_  = 32;
constexpr int SS_  = 512;
constexpr int NTOK = NB_ * SS_;        // 16384 rows
constexpr int OBS_ = 256;
constexpr int ACT_ = 32;
constexpr int HID_ = 2048;
constexpr int LAT_ = 256;
constexpr int KC_  = 4096;             // codebook entries
constexpr int CIN_ = OBS_ + ACT_;      // 288 true input dim
constexpr int CINP = 320;              // zero-padded to a multiple of 64

// round-to-nearest-even float -> bf16 bits
__device__ __forceinline__ unsigned short f2bf(float f) {
  union { float f; unsigned u; } v; v.f = f;
  unsigned u = v.u;
  u += 0x7FFFu + ((u >> 16) & 1u);
  return (unsigned short)(u >> 16);
}

// Async DMA 16 bytes global -> LDS (no VGPR round trip, tracked by ASYNCcnt).
// LDS operand is the per-lane LDS byte offset = low 32 bits of the generic
// shared-memory address (ISA 10.2).
__device__ __forceinline__ void async_copy_b128(const unsigned short* g,
                                                const unsigned short* l) {
  const unsigned loff = (unsigned)(uintptr_t)l;
  asm volatile("global_load_async_to_lds_b128 %0, %1, off"
               :: "v"(loff), "v"(g)
               : "memory");
}
__device__ __forceinline__ void wait_async_le10() {
  asm volatile("s_wait_asynccnt 0xa" ::: "memory");
}
__device__ __forceinline__ void wait_async_0() {
  asm volatile("s_wait_asynccnt 0x0" ::: "memory");
}

// A-matrix fragment, 16-bit 16x32 (ISA 7.12.2): lane L -> row = L&15,
// k elements { h*8 .. h*8+7 , 16+h*8 .. 16+h*8+7 }, h = L>>4.
__device__ __forceinline__ v16bf load_a_frag(const unsigned short* base, int ld,
                                             int row0, int k0) {
  const int lane = threadIdx.x & 31;
  const int m = lane & 15, half = lane >> 4;
  const unsigned short* p = base + (size_t)(row0 + m) * ld + k0 + half * 8;
  union { v16bf v; uint4 q[2]; } u;
  u.q[0] = *reinterpret_cast<const uint4*>(p);
  u.q[1] = *reinterpret_cast<const uint4*>(p + 16);
  return u.v;
}

// B-matrix fragment, 16-bit 32x16: lane L -> col = L&15,
// k elements { h*8 .. h*8+7 , 16+h*8 .. 16+h*8+7 }, from storage Bt[col][k].
__device__ __forceinline__ v16bf load_b_frag(const unsigned short* base, int ld,
                                             int col0, int k0) {
  const int lane = threadIdx.x & 31;
  const int n = lane & 15, half = lane >> 4;
  const unsigned short* p = base + (size_t)(col0 + n) * ld + k0 + half * 8;
  union { v16bf v; uint4 q[2]; } u;
  u.q[0] = *reinterpret_cast<const uint4*>(p);
  u.q[1] = *reinterpret_cast<const uint4*>(p + 16);
  return u.v;
}

// ---------------------------------------------------------------------------
// Tiled GEMM: O = act(A[M,K](bf16) * Wt[N,K]^T(bf16) + bias), f32 accumulate.
// Block tile 64x256, 8 waves (2x4), each wave 32x64 (2x4 WMMA tiles).
// k-slab = 64 (2 WMMA k-steps); A/B slabs double-buffered in LDS, filled by
// async global->LDS DMA. Steady-state iteration is branch-free:
//   stage(next) -> s_wait_asynccnt 10 -> barrier -> 16 WMMAs -> barrier.
// (10 async ops per wave per slab: 2 for A, 8 for B; in-order completion
//  means waiting to <=10 retires exactly the previous slab.)
// ---------------------------------------------------------------------------
__global__ __launch_bounds__(256) void k_gemm(
    const unsigned short* __restrict__ A, const unsigned short* __restrict__ Wt,
    const float* __restrict__ bias, unsigned short* __restrict__ Obf,
    float* __restrict__ Of32, int M, int N, int K, int relu) {
  __shared__ __align__(16) unsigned short As[2][64 * 64];    // 2 x 8 KB
  __shared__ __align__(16) unsigned short Bs[2][256 * 64];   // 2 x 32 KB
  const int tid  = threadIdx.x;
  const int wave = tid >> 5;
  const int lane = tid & 31;
  const int bm = blockIdx.y * 64;
  const int bn = blockIdx.x * 256;
  const int wm = (wave >> 2) * 32;   // 0 / 32
  const int wn = (wave & 3) * 64;    // 0 / 64 / 128 / 192

  // issue one k-slab (64 wide) of A (64x64) and B (256x64) into LDS buffer b.
  // A: 512 16B-chunks -> 2/thread; B: 2048 chunks -> 8/thread (one full row).
  auto stage = [&](int k0, int b) {
#pragma unroll
    for (int j = 0; j < 2; ++j) {
      const int ch = tid * 2 + j;
      const int r = ch >> 3, c = (ch & 7) * 8;
      async_copy_b128(&A[(size_t)(bm + r) * K + k0 + c], &As[b][r * 64 + c]);
    }
#pragma unroll
    for (int j = 0; j < 8; ++j) {
      const int c = j * 8;
      async_copy_b128(&Wt[(size_t)(bn + tid) * K + k0 + c], &Bs[b][tid * 64 + c]);
    }
  };

  const v8f vzero = {0.f, 0.f, 0.f, 0.f, 0.f, 0.f, 0.f, 0.f};
  v8f acc[2][4];
#pragma unroll
  for (int i = 0; i < 2; ++i)
#pragma unroll
    for (int j = 0; j < 4; ++j) acc[i][j] = vzero;

  auto compute = [&](int b) {
    const unsigned short* As_ = &As[b][0];
    const unsigned short* Bs_ = &Bs[b][0];
#pragma unroll
    for (int kk = 0; kk < 2; ++kk) {
      v16bf a0 = load_a_frag(As_, 64, wm, kk * 32);
      v16bf a1 = load_a_frag(As_, 64, wm + 16, kk * 32);
#pragma unroll
      for (int j = 0; j < 4; ++j) {
        v16bf b0 = load_b_frag(Bs_, 64, wn + j * 16, kk * 32);
        acc[0][j] = __builtin_amdgcn_wmma_f32_16x16x32_bf16(false, a0, false, b0, (short)0, acc[0][j], false, false);
        acc[1][j] = __builtin_amdgcn_wmma_f32_16x16x32_bf16(false, a1, false, b0, (short)0, acc[1][j], false, false);
      }
    }
  };

  stage(0, 0);
  int buf = 0;
  int k0 = 0;
  for (; k0 + 64 < K; k0 += 64) {        // branch-free steady state
    stage(k0 + 64, buf ^ 1);             // DMA next slab into other buf
    __builtin_prefetch(&A[(size_t)(bm + (tid >> 2)) * K + k0 + 128], 0, 1);
    __builtin_prefetch(&Wt[(size_t)(bn + tid) * K + k0 + 128], 0, 1);
    wait_async_le10();                   // oldest 10 (current slab) done
    __syncthreads();                     // all waves' slab data visible
    compute(buf);
    __syncthreads();                     // reads done before buf reuse
    buf ^= 1;
  }
  wait_async_0();                        // final slab
  __syncthreads();
  compute(buf);

  // Epilogue. C layout: VGPR v -> row v (lanes 0-15) / row v+8 (lanes 16-31),
  // col = lane&15.
  const int cit  = lane & 15;
  const int half = lane >> 4;
#pragma unroll
  for (int i = 0; i < 2; ++i) {
#pragma unroll
    for (int j = 0; j < 4; ++j) {
      const int row0 = bm + wm + i * 16 + half * 8;
      const int col  = bn + wn + j * 16 + cit;
      const float bv = bias ? bias[col] : 0.f;
#pragma unroll
      for (int v = 0; v < 8; ++v) {
        float r = acc[i][j][v] + bv;
        if (relu) r = fmaxf(r, 0.f);
        const size_t o = (size_t)(row0 + v) * N + col;
        if (Obf)  Obf[o]  = f2bf(r);
        if (Of32) Of32[o] = r;
      }
    }
  }
}

// ---------------------------------------------------------------------------
// Fused nearest-code search: argmin_k ( ||e_k||^2 - 2 * <x, e_k> ).
// One wave owns 16 rows; loops over all 4096 codes in 16-col WMMA tiles,
// per-lane running (min,argmin), final 16-lane shfl_xor butterfly.
// ---------------------------------------------------------------------------
__global__ __launch_bounds__(256) void k_argmin(
    const unsigned short* __restrict__ latbf, const unsigned short* __restrict__ embbf,
    const float* __restrict__ embsq, int* __restrict__ tokens) {
  const int wave = blockIdx.x * 8 + (threadIdx.x >> 5);
  const int lane = threadIdx.x & 31;
  const int row0 = wave * 16;
  const v8f vzero = {0.f, 0.f, 0.f, 0.f, 0.f, 0.f, 0.f, 0.f};

  v16bf a[8];
#pragma unroll
  for (int kk = 0; kk < 8; ++kk) a[kk] = load_a_frag(latbf, LAT_, row0, kk * 32);

  float bestV[8];
  int   bestI[8];
#pragma unroll
  for (int v = 0; v < 8; ++v) { bestV[v] = 3.4e38f; bestI[v] = 0; }

  const int cit = lane & 15;
  for (int c0 = 0; c0 < KC_; c0 += 16) {
    v8f acc = vzero;
#pragma unroll
    for (int kk = 0; kk < 8; ++kk) {
      v16bf b = load_b_frag(embbf, LAT_, c0, kk * 32);
      acc = __builtin_amdgcn_wmma_f32_16x16x32_bf16(false, a[kk], false, b, (short)0, acc, false, false);
    }
    const int gcol = c0 + cit;
    const float es = embsq[gcol];
#pragma unroll
    for (int v = 0; v < 8; ++v) {
      const float d = es - 2.0f * acc[v];
      if (d < bestV[v]) { bestV[v] = d; bestI[v] = gcol; }
    }
  }

  const int half = lane >> 4;
#pragma unroll
  for (int v = 0; v < 8; ++v) {
    float bv = bestV[v];
    int   bi = bestI[v];
#pragma unroll
    for (int off = 8; off >= 1; off >>= 1) {
      const float ov = __shfl_xor(bv, off, 16);
      const int   oi = __shfl_xor(bi, off, 16);
      if (ov < bv || (ov == bv && oi < bi)) { bv = ov; bi = oi; }
    }
    if (cit == 0) tokens[row0 + half * 8 + v] = bi;  // first-index tie break
  }
}

// ---------------------------------------------------------------------------
// Support kernels
// ---------------------------------------------------------------------------
__global__ __launch_bounds__(256) void k_pack_xenc(
    const float* __restrict__ obs, const float* __restrict__ actions,
    unsigned short* __restrict__ X) {
  const size_t i = (size_t)blockIdx.x * blockDim.x + threadIdx.x;
  if (i >= (size_t)NTOK * CINP) return;
  const int n = (int)(i / CINP), c = (int)(i % CINP);
  float v = 0.f;
  if (c < OBS_)      v = obs[(size_t)n * OBS_ + c];
  else if (c < CIN_) v = actions[(size_t)n * ACT_ + (c - OBS_)];
  X[i] = f2bf(v);
}

// W[K,N](f32) -> Wt[N,Kp](bf16), zero-padded for k >= K
__global__ __launch_bounds__(256) void k_transpose(
    const float* __restrict__ W, unsigned short* __restrict__ Wt,
    int K, int N, int Kp) {
  const size_t i = (size_t)blockIdx.x * blockDim.x + threadIdx.x;
  if (i >= (size_t)N * Kp) return;
  const int n = (int)(i / Kp), k = (int)(i % Kp);
  Wt[i] = (k < K) ? f2bf(W[(size_t)k * N + n]) : (unsigned short)0;
}

__global__ __launch_bounds__(256) void k_emb_prep(
    const float* __restrict__ emb, unsigned short* __restrict__ embbf,
    float* __restrict__ embsq) {
  const int r = blockIdx.x, c = threadIdx.x;  // 4096 blocks x 256 threads
  const float v = emb[(size_t)r * LAT_ + c];
  embbf[(size_t)r * LAT_ + c] = f2bf(v);
  float s = v * v;
#pragma unroll
  for (int off = 16; off >= 1; off >>= 1) s += __shfl_xor(s, off, 32);
  __shared__ float red[8];
  if ((c & 31) == 0) red[c >> 5] = s;
  __syncthreads();
  if (c == 0) {
    float t = 0.f;
#pragma unroll
    for (int i = 0; i < 8; ++i) t += red[i];
    embsq[r] = t;
  }
}

__global__ __launch_bounds__(256) void k_gather_pack(
    const int* __restrict__ tokens, const float* __restrict__ emb,
    const float* __restrict__ actions, const float* __restrict__ lat,
    unsigned short* __restrict__ Xdec, float* __restrict__ out_qste,
    float* __restrict__ out_tok, float* __restrict__ accum) {
  const size_t i = (size_t)blockIdx.x * blockDim.x + threadIdx.x;
  float sq = 0.f;
  if (i < (size_t)NTOK * CINP) {
    const int n = (int)(i / CINP), c = (int)(i % CINP);
    float val = 0.f;
    if (c < LAT_) {
      const int idx = tokens[n];
      val = emb[(size_t)idx * LAT_ + c];
      out_qste[(size_t)n * LAT_ + c] = val;   // STE forward value == quantized
      const float d = lat[(size_t)n * LAT_ + c] - val;
      sq = d * d;
    } else if (c < CIN_) {
      val = actions[(size_t)n * ACT_ + (c - LAT_)];
      if (c == CIN_ - 1) out_tok[n] = (float)tokens[n];
    }
    Xdec[i] = f2bf(val);
  }
#pragma unroll
  for (int off = 16; off >= 1; off >>= 1) sq += __shfl_xor(sq, off, 32);
  if ((threadIdx.x & 31) == 0) atomicAdd(&accum[0], sq);
}

__global__ __launch_bounds__(256) void k_recon_loss(
    const float* __restrict__ recon, const float* __restrict__ obs,
    float* __restrict__ accum) {
  const size_t i = (size_t)blockIdx.x * blockDim.x + threadIdx.x;
  float sq = 0.f;
  if (i < (size_t)NTOK * OBS_) {
    const float d = recon[i] - obs[i];
    sq = d * d;
  }
#pragma unroll
  for (int off = 16; off >= 1; off >>= 1) sq += __shfl_xor(sq, off, 32);
  if ((threadIdx.x & 31) == 0) atomicAdd(&accum[1], sq);
}

__global__ void k_init(float* __restrict__ accum) {
  if (threadIdx.x < 8) accum[threadIdx.x] = 0.f;
}

__global__ void k_final(const float* __restrict__ accum, float* __restrict__ out_scal) {
  const float msq    = accum[0] / (float)((size_t)NTOK * LAT_);
  const float commit = 0.25f * msq;        // COMMIT * mean(diff^2)
  const float codeb  = msq;
  const float tq     = commit + codeb;
  const float rl     = accum[1] / (float)((size_t)NTOK * OBS_);
  out_scal[0] = rl;
  out_scal[1] = commit;
  out_scal[2] = codeb;
  out_scal[3] = tq;
  out_scal[4] = rl + tq;
}

// ---------------------------------------------------------------------------
// Launch
// ---------------------------------------------------------------------------
extern "C" void kernel_launch(void* const* d_in, const int* in_sizes, int n_in,
                              void* d_out, int out_size, void* d_ws, size_t ws_size,
                              hipStream_t stream) {
  (void)in_sizes; (void)n_in; (void)out_size; (void)ws_size;
  const float* obs     = (const float*)d_in[0];
  const float* actions = (const float*)d_in[1];
  const float* enc_w1  = (const float*)d_in[2];
  const float* enc_b1  = (const float*)d_in[3];
  const float* enc_w2  = (const float*)d_in[4];
  const float* enc_b2  = (const float*)d_in[5];
  const float* enc_w3  = (const float*)d_in[6];
  const float* enc_b3  = (const float*)d_in[7];
  const float* emb     = (const float*)d_in[8];
  const float* dec_w1  = (const float*)d_in[9];
  const float* dec_b1  = (const float*)d_in[10];
  const float* dec_w2  = (const float*)d_in[11];
  const float* dec_b2  = (const float*)d_in[12];
  const float* dec_w3  = (const float*)d_in[13];
  const float* dec_b3  = (const float*)d_in[14];

  // d_out layout (floats): recon | tokens | quantized_ste | latents | 5 scalars
  float* out        = (float*)d_out;
  float* out_recon  = out;
  float* out_tok    = out_recon + (size_t)NTOK * OBS_;
  float* out_qste   = out_tok + NTOK;
  float* out_lat    = out_qste + (size_t)NTOK * LAT_;
  float* out_scal   = out_lat + (size_t)NTOK * LAT_;

  // workspace carve (256B aligned)
  char* w = (char*)d_ws;
  size_t off = 0;
  auto alloc = [&](size_t bytes) -> void* {
    void* p = w + off;
    off = (off + bytes + 255) & ~(size_t)255;
    return p;
  };
  unsigned short* Xenc  = (unsigned short*)alloc((size_t)NTOK * CINP * 2);
  unsigned short* h1    = (unsigned short*)alloc((size_t)NTOK * HID_ * 2);
  unsigned short* h2    = (unsigned short*)alloc((size_t)NTOK * HID_ * 2);
  unsigned short* latbf = (unsigned short*)alloc((size_t)NTOK * LAT_ * 2);
  unsigned short* Xdec  = (unsigned short*)alloc((size_t)NTOK * CINP * 2);
  int*            toks  = (int*)alloc((size_t)NTOK * 4);
  unsigned short* w1te  = (unsigned short*)alloc((size_t)CINP * HID_ * 2);
  unsigned short* w2te  = (unsigned short*)alloc((size_t)HID_ * HID_ * 2);
  unsigned short* w3te  = (unsigned short*)alloc((size_t)HID_ * LAT_ * 2);
  unsigned short* w1td  = (unsigned short*)alloc((size_t)CINP * HID_ * 2);
  unsigned short* w2td  = (unsigned short*)alloc((size_t)HID_ * HID_ * 2);
  unsigned short* w3td  = (unsigned short*)alloc((size_t)HID_ * OBS_ * 2);
  unsigned short* embbf = (unsigned short*)alloc((size_t)KC_ * LAT_ * 2);
  float*          embsq = (float*)alloc((size_t)KC_ * 4);
  float*          accum = (float*)alloc(8 * 4);

  auto gemm = [&](const unsigned short* A, const unsigned short* Wt, const float* bias,
                  unsigned short* Obf, float* Of32, int M, int N, int K, int relu) {
    dim3 grid(N / 256, M / 64);
    k_gemm<<<grid, 256, 0, stream>>>(A, Wt, bias, Obf, Of32, M, N, K, relu);
  };
  auto tr = [&](const float* W, unsigned short* Wt, int K, int N, int Kp) {
    const size_t total = (size_t)N * Kp;
    k_transpose<<<(total + 255) / 256, 256, 0, stream>>>(W, Wt, K, N, Kp);
  };

  k_init<<<1, 32, 0, stream>>>(accum);
  k_pack_xenc<<<((size_t)NTOK * CINP + 255) / 256, 256, 0, stream>>>(obs, actions, Xenc);
  tr(enc_w1, w1te, CIN_, HID_, CINP);
  tr(enc_w2, w2te, HID_, HID_, HID_);
  tr(enc_w3, w3te, HID_, LAT_, HID_);
  tr(dec_w1, w1td, CIN_, HID_, CINP);
  tr(dec_w2, w2td, HID_, HID_, HID_);
  tr(dec_w3, w3td, HID_, OBS_, HID_);
  k_emb_prep<<<KC_, 256, 0, stream>>>(emb, embbf, embsq);

  // encoder
  gemm(Xenc, w1te, enc_b1, h1, nullptr, NTOK, HID_, CINP, 1);
  gemm(h1,   w2te, enc_b2, h2, nullptr, NTOK, HID_, HID_, 1);
  gemm(h2,   w3te, enc_b3, latbf, out_lat, NTOK, LAT_, HID_, 0);

  // quantize
  k_argmin<<<NTOK / (16 * 8), 256, 0, stream>>>(latbf, embbf, embsq, toks);
  k_gather_pack<<<((size_t)NTOK * CINP + 255) / 256, 256, 0, stream>>>(
      toks, emb, actions, out_lat, Xdec, out_qste, out_tok, accum);

  // decoder
  gemm(Xdec, w1td, dec_b1, h1, nullptr, NTOK, HID_, CINP, 1);
  gemm(h1,   w2td, dec_b2, h2, nullptr, NTOK, HID_, HID_, 1);
  gemm(h2,   w3td, dec_b3, nullptr, out_recon, NTOK, OBS_, HID_, 0);

  // losses
  k_recon_loss<<<((size_t)NTOK * OBS_ + 255) / 256, 256, 0, stream>>>(out_recon, obs, accum);
  k_final<<<1, 1, 0, stream>>>(accum, out_scal);
}
